// InteractingVSSBlock_10411000725623
// MI455X (gfx1250) — matrix operand order repaired and checked by reading, not compiled, not run on hardware
//
#include <hip/hip_runtime.h>
#include <hip/hip_bf16.h>

// ---------------------------------------------------------------------------
// Shapes (compile-time constants from the reference)
// ---------------------------------------------------------------------------
#define Bc   8
#define Hc   64
#define Wc   64
#define Cc   64
#define DIc  128
#define Lc   4096            // H*W
#define Mc   32768           // B*L rows
#define Kdir 4
#define Rr   4
#define OUT_HALF 2097152     // Mc*Cc, offset of second tuple element in d_out

typedef __attribute__((ext_vector_type(16))) _Float16 v16h;
typedef __attribute__((ext_vector_type(8)))  float    v8f;

__device__ __forceinline__ float silu_f(float x) { return x / (1.f + __expf(-x)); }
__device__ __forceinline__ float gelu_f(float x) { return 0.5f * x * (1.f + erff(x * 0.70710678118f)); }
__device__ __forceinline__ float softplus_f(float x) { return (x > 20.f) ? x : log1pf(__expf(x)); }

// ---------------------------------------------------------------------------
// fp32 -> fp16 conversion (weights)
// ---------------------------------------------------------------------------
__global__ void f32_to_f16(const float* __restrict__ s, _Float16* __restrict__ d, int n) {
    int i = blockIdx.x * blockDim.x + threadIdx.x;
    if (i < n) d[i] = (_Float16)s[i];
}

__global__ void zerof(float* __restrict__ p, int n) {
    int i = blockIdx.x * blockDim.x + threadIdx.x;
    if (i < n) p[i] = 0.f;
}

// ---------------------------------------------------------------------------
// LayerNorm over 64 cols, one wave32 per row, fp16 output
// ---------------------------------------------------------------------------
__global__ __launch_bounds__(256) void ln64_f16(const float* __restrict__ x,
                                                const float* __restrict__ w,
                                                const float* __restrict__ b,
                                                _Float16* __restrict__ out, int rows) {
    int wave = (int)((blockIdx.x * blockDim.x + threadIdx.x) >> 5);
    int lane = threadIdx.x & 31;
    if (wave >= rows) return;
    const float* xr = x + (size_t)wave * 64;
    float v0 = xr[lane], v1 = xr[lane + 32];
    float s = v0 + v1;
    #pragma unroll
    for (int i = 16; i > 0; i >>= 1) s += __shfl_xor(s, i, 32);
    float mu = s * (1.f / 64.f);
    float d0 = v0 - mu, d1 = v1 - mu;
    float q = d0 * d0 + d1 * d1;
    #pragma unroll
    for (int i = 16; i > 0; i >>= 1) q += __shfl_xor(q, i, 32);
    float rstd = rsqrtf(q * (1.f / 64.f) + 1e-5f);
    out[(size_t)wave * 64 + lane]      = (_Float16)(d0 * rstd * w[lane]      + b[lane]);
    out[(size_t)wave * 64 + lane + 32] = (_Float16)(d1 * rstd * w[lane + 32] + b[lane + 32]);
}

// ---------------------------------------------------------------------------
// out-norm: LayerNorm over 128 cols, multiply by silu(z), fp16 output
// ---------------------------------------------------------------------------
__global__ __launch_bounds__(256) void outln128_mulz(const float* __restrict__ y,
                                                     const float* __restrict__ w,
                                                     const float* __restrict__ b,
                                                     const float* __restrict__ z,
                                                     _Float16* __restrict__ g, int rows) {
    int wave = (int)((blockIdx.x * blockDim.x + threadIdx.x) >> 5);
    int lane = threadIdx.x & 31;
    if (wave >= rows) return;
    const float* yr = y + (size_t)wave * 128;
    float v[4];
    float s = 0.f;
    #pragma unroll
    for (int j = 0; j < 4; ++j) { v[j] = yr[lane + 32 * j]; s += v[j]; }
    #pragma unroll
    for (int i = 16; i > 0; i >>= 1) s += __shfl_xor(s, i, 32);
    float mu = s * (1.f / 128.f);
    float q = 0.f;
    #pragma unroll
    for (int j = 0; j < 4; ++j) { v[j] -= mu; q += v[j] * v[j]; }
    #pragma unroll
    for (int i = 16; i > 0; i >>= 1) q += __shfl_xor(q, i, 32);
    float rstd = rsqrtf(q * (1.f / 128.f) + 1e-5f);
    const float* zr = z + (size_t)wave * 128;
    #pragma unroll
    for (int j = 0; j < 4; ++j) {
        int c = lane + 32 * j;
        g[(size_t)wave * 128 + c] = (_Float16)((v[j] * rstd * w[c] + b[c]) * zr[c]);
    }
}

// ---------------------------------------------------------------------------
// WMMA GEMM: D[M,N] = A[M,K] @ Bw[N,K]^T with fused epilogue.
// One wave computes a 16 x (16*NT) stripe: the A tile (1KB) is loaded once per
// k-step and reused across NT column tiles (NT back-to-back v_wmma per step),
// amortizing operand bandwidth 4x and quartering wavefront count.
//   EPI 0: in_proj split  -> xc (NCHW f32) cols<128, silu(z) cols>=128
//   EPI 1: out_proj        -> out0 = val + resid        (f32)
//   EPI 2: fc1             -> out16 = gelu(val + bias)  (f16)
//   EPI 3: fc2 branch 0    -> out0 = val + bias + resid (f32)
//   EPI 4: fc2 branch 1    -> s = val+bias+resid+aux; d_out[idx]=s; d_out[idx+OUT_HALF]=s
// ---------------------------------------------------------------------------
template <int KDIM, int NT, int EPI>
__global__ __launch_bounds__(32) void wmma_gemm(const _Float16* __restrict__ A,
                                                const _Float16* __restrict__ Bw,
                                                const float* __restrict__ bias,
                                                const float* __restrict__ resid,
                                                const float* __restrict__ aux,
                                                float* __restrict__ out0,
                                                float* __restrict__ out1,
                                                _Float16* __restrict__ out16) {
    const int lane = threadIdx.x;
    const int half = lane >> 4;            // 0 or 1
    const int lidx = lane & 15;
    const int mrow = blockIdx.x * 16 + lidx;       // A row owned by this lane
    const int nbase = blockIdx.y * (16 * NT);      // first output col of this wave

    v8f acc[NT] = {};
    const _Float16* arow = A + (size_t)mrow * KDIM;

    #pragma unroll
    for (int k0 = 0; k0 < KDIM; k0 += 32) {
        if (k0 + 32 < KDIM) {
            __builtin_prefetch(arow + k0 + 32, 0, 3);   // global_prefetch_b8
        }
        // A 16x32 f16 layout: lane(0-15)=row, K interleaved by half-wave:
        //   K(v,half) = (v>=4 ? 16 : 0) + half*8 + 2*(v&3)
        v16h a;
        #pragma unroll
        for (int v = 0; v < 8; ++v) {
            int kk = k0 + ((v & 4) ? 16 : 0) + half * 8 + 2 * (v & 3);
            a[2 * v]     = arow[kk];
            a[2 * v + 1] = arow[kk + 1];
        }
        // B 32x16 f16 layout: lane gives col, K interleaved by half-wave like A:
        //   K(e,half) = (e>=8 ? 16 : 0) + half*8 + (e&7).  NT tiles back-to-back reuse `a`.
        #pragma unroll
        for (int t = 0; t < NT; ++t) {
            const _Float16* brow =
                Bw + (size_t)(nbase + t * 16 + lidx) * KDIM + half * 8 + k0;
            v16h b;
            #pragma unroll
            for (int e = 0; e < 16; ++e) b[e] = brow[e < 8 ? e : e + 8];
            acc[t] = __builtin_amdgcn_wmma_f32_16x16x32_f16(false, a, false, b,
                                                            (short)0, acc[t],
                                                            false, false);
        }
    }

    // D layout: VGPR v -> row (v + 8*half), lane&15 -> col
    #pragma unroll
    for (int t = 0; t < NT; ++t) {
        #pragma unroll
        for (int v = 0; v < 8; ++v) {
            int row = blockIdx.x * 16 + v + half * 8;
            int col = nbase + t * 16 + lidx;
            float val = acc[t][v];
            if (EPI == 0) {                       // in_proj: split xc (NCHW) / silu(z)
                if (col < DIc) {
                    int bb = row >> 12, hw = row & 4095;
                    out0[((size_t)(bb * DIc + col)) * Lc + hw] = val;
                } else {
                    out1[(size_t)row * DIc + (col - DIc)] = silu_f(val);
                }
            } else if (EPI == 1) {                // out_proj + residual
                size_t idx = (size_t)row * Cc + col;
                out0[idx] = val + resid[idx];
            } else if (EPI == 2) {                // fc1 + bias + gelu -> f16
                out16[(size_t)row * 256 + col] = (_Float16)gelu_f(val + bias[col]);
            } else if (EPI == 3) {                // fc2 (branch 0)
                size_t idx = (size_t)row * Cc + col;
                out0[idx] = val + bias[col] + resid[idx];
            } else {                              // fc2 (branch 1): sum branches, dup output
                size_t idx = (size_t)row * Cc + col;
                float s = val + bias[col] + resid[idx] + aux[idx];
                out0[idx] = s;
                out0[idx + OUT_HALF] = s;
            }
        }
    }
}

// ---------------------------------------------------------------------------
// depthwise 3x3 conv (pad 1) + SiLU, writing both scan layouts:
//   x0T[b][h*W+w][d]  (row-major stream)   x1T[b][w*H+h][d] (col-major stream)
// ---------------------------------------------------------------------------
__global__ __launch_bounds__(256) void conv_silu(const float* __restrict__ xc,
                                                 const float* __restrict__ cw,
                                                 float* __restrict__ x0T,
                                                 float* __restrict__ x1T) {
    int t = blockIdx.x * blockDim.x + threadIdx.x;     // < B*DI*L
    int l = t & 4095;
    int d = (t >> 12) & 127;
    int b = t >> 19;
    int h = l >> 6, w = l & 63;
    const float* base = xc + ((size_t)(b * DIc + d)) * Lc;
    const float* wp = cw + d * 9;
    float s = 0.f;
    #pragma unroll
    for (int kh = 0; kh < 3; ++kh) {
        int hh = h + kh - 1;
        if ((unsigned)hh < 64u) {
            #pragma unroll
            for (int kw = 0; kw < 3; ++kw) {
                int ww = w + kw - 1;
                if ((unsigned)ww < 64u) s += wp[kh * 3 + kw] * base[hh * 64 + ww];
            }
        }
    }
    s = silu_f(s);
    x0T[((size_t)b * Lc + l) * DIc + d] = s;
    x1T[((size_t)b * Lc + (w * 64 + h)) * DIc + d] = s;
}

// ---------------------------------------------------------------------------
// x_dbl[b][k][c][l] = sum_d x_proj_w[k][c][d] * xs[k][d][l]  (c = 0..5)
// direction k selects stream + reversal by pure index transform.
// ---------------------------------------------------------------------------
__global__ __launch_bounds__(256) void xdbl_kernel(const float* __restrict__ x0T,
                                                   const float* __restrict__ x1T,
                                                   const float* __restrict__ xpw,
                                                   float* __restrict__ xdbl) {
    __shared__ float wsm[6 * DIc];
    int k = blockIdx.y, b = blockIdx.z;
    for (int i = threadIdx.x; i < 6 * DIc; i += 256) wsm[i] = xpw[k * 6 * DIc + i];
    __syncthreads();
    int l = blockIdx.x * 256 + threadIdx.x;
    int ix = (k & 2) ? (4095 - l) : l;
    const float* src = ((k & 1) ? x1T : x0T) + ((size_t)b * Lc + ix) * DIc;
    float a0 = 0, a1 = 0, a2 = 0, a3 = 0, a4 = 0, a5 = 0;
    #pragma unroll 4
    for (int d = 0; d < DIc; ++d) {
        float xv = src[d];
        a0 += wsm[d] * xv;           a1 += wsm[DIc + d] * xv;
        a2 += wsm[2 * DIc + d] * xv; a3 += wsm[3 * DIc + d] * xv;
        a4 += wsm[4 * DIc + d] * xv; a5 += wsm[5 * DIc + d] * xv;
    }
    size_t base = ((size_t)(b * Kdir + k)) * 6 * Lc + l;
    xdbl[base]          = a0; xdbl[base + Lc]     = a1;
    xdbl[base + 2 * Lc] = a2; xdbl[base + 3 * Lc] = a3;
    xdbl[base + 4 * Lc] = a4; xdbl[base + 5 * Lc] = a5;
}

// ---------------------------------------------------------------------------
// Selective scan (N=1 state): 128 threads = one channel each, sequential over L.
// Direction merge fused via global_atomic_add_f32 into canonical layout
// ycomb[b][h*W+w][d].
// ---------------------------------------------------------------------------
__global__ __launch_bounds__(128) void scan_kernel(const float* __restrict__ xdbl,
                                                   const float* __restrict__ x0T,
                                                   const float* __restrict__ x1T,
                                                   const float* __restrict__ dtw,
                                                   const float* __restrict__ dtb,
                                                   const float* __restrict__ alog,
                                                   const float* __restrict__ dsv,
                                                   float* __restrict__ ycomb) {
    int k = blockIdx.x & 3, b = blockIdx.x >> 2;
    int d = threadIdx.x;
    int kd = k * DIc + d;
    float w0 = dtw[kd * 4 + 0], w1 = dtw[kd * 4 + 1];
    float w2 = dtw[kd * 4 + 2], w3 = dtw[kd * 4 + 3];
    float bb = dtb[kd];
    float Aa = -__expf(alog[kd]);
    float Dv = dsv[kd];
    const float* cd = xdbl + ((size_t)(b * Kdir + k)) * 6 * Lc;
    const float* src = ((k & 1) ? x1T : x0T) + (size_t)b * Lc * DIc;
    float hstate = 0.f;
    for (int l = 0; l < Lc; ++l) {
        int ix = (k & 2) ? (4095 - l) : l;          // index in the stream's own space
        float xv = src[(size_t)ix * DIc + d];
        float u = w0 * cd[l] + w1 * cd[Lc + l] + w2 * cd[2 * Lc + l] + w3 * cd[3 * Lc + l] + bb;
        float delta = softplus_f(u);
        float Bv = cd[4 * Lc + l], Cv = cd[5 * Lc + l];
        float a = __expf(delta * Aa);
        hstate = hstate * a + delta * Bv * xv;
        float y = hstate * Cv + Dv * xv;
        // canonical spatial position (merge of the 4 directions):
        int p = (k & 1) ? ((ix & 63) * 64 + (ix >> 6)) : ix;
        atomicAdd(&ycomb[((size_t)b * Lc + p) * DIc + d], y);
    }
}

// ---------------------------------------------------------------------------
// Host-side launch
// ---------------------------------------------------------------------------
extern "C" void kernel_launch(void* const* d_in, const int* in_sizes, int n_in,
                              void* d_out, int out_size, void* d_ws, size_t ws_size,
                              hipStream_t stream) {
    (void)in_sizes; (void)n_in; (void)out_size; (void)ws_size;

    const float* X[2]  = {(const float*)d_in[0], (const float*)d_in[1]};
    const float* n1w   = (const float*)d_in[2];
    const float* n1b   = (const float*)d_in[3];
    const float* inpw  = (const float*)d_in[4];   // (2,256,64)
    const float* convw = (const float*)d_in[5];   // (2,128,1,3,3)
    const float* xpw   = (const float*)d_in[6];   // (2,4,6,128)
    const float* dtw   = (const float*)d_in[7];   // (2,4,128,4)
    const float* dtb   = (const float*)d_in[8];   // (2,4,128)
    const float* alog  = (const float*)d_in[9];   // (2,512,1)
    const float* dsv   = (const float*)d_in[10];  // (2,512)
    const float* onw   = (const float*)d_in[11];
    const float* onb   = (const float*)d_in[12];
    const float* opw   = (const float*)d_in[13];  // (2,64,128)
    const float* n2w   = (const float*)d_in[14];
    const float* n2b   = (const float*)d_in[15];
    const float* fc1w  = (const float*)d_in[16];  // (2,256,64)
    const float* fc1b  = (const float*)d_in[17];
    const float* fc2w  = (const float*)d_in[18];  // (2,64,256)
    const float* fc2b  = (const float*)d_in[19];
    float* out = (float*)d_out;

    // ---- workspace carving (buffers reused across the two serialized branches)
    char* p = (char*)d_ws;
    auto alloc = [&](size_t bytes) -> void* {
        void* r = (void*)p;
        p += (bytes + 255) & ~(size_t)255;
        return r;
    };
    _Float16* h16    = (_Float16*)alloc((size_t)Mc * 64 * 2);
    _Float16* inp16  = (_Float16*)alloc(256 * 64 * 2);
    _Float16* outp16 = (_Float16*)alloc(64 * 128 * 2);
    _Float16* fc1w16 = (_Float16*)alloc(256 * 64 * 2);
    _Float16* fc2w16 = (_Float16*)alloc(64 * 256 * 2);
    float*    xc     = (float*)alloc((size_t)Bc * DIc * Lc * 4);
    float*    zbuf   = (float*)alloc((size_t)Mc * DIc * 4);
    float*    x0T    = (float*)alloc((size_t)Bc * Lc * DIc * 4);
    float*    x1T    = (float*)alloc((size_t)Bc * Lc * DIc * 4);
    float*    xdbl   = (float*)alloc((size_t)Bc * Kdir * 6 * Lc * 4);
    float*    ycomb  = (float*)alloc((size_t)Mc * DIc * 4);
    _Float16* g16    = (_Float16*)alloc((size_t)Mc * DIc * 2);
    float*    xres   = (float*)alloc((size_t)Mc * 64 * 4);
    _Float16* h2_16  = (_Float16*)alloc((size_t)Mc * 64 * 2);
    _Float16* a16    = (_Float16*)alloc((size_t)Mc * 256 * 2);
    float*    bout   = (float*)alloc((size_t)Mc * 64 * 4);

    for (int br = 0; br < 2; ++br) {
        // weight conversion fp32 -> fp16
        f32_to_f16<<<(256 * 64 + 255) / 256, 256, 0, stream>>>(inpw + br * 256 * 64, inp16, 256 * 64);
        f32_to_f16<<<(64 * 128 + 255) / 256, 256, 0, stream>>>(opw + br * 64 * 128, outp16, 64 * 128);
        f32_to_f16<<<(256 * 64 + 255) / 256, 256, 0, stream>>>(fc1w + br * 256 * 64, fc1w16, 256 * 64);
        f32_to_f16<<<(64 * 256 + 255) / 256, 256, 0, stream>>>(fc2w + br * 64 * 256, fc2w16, 64 * 256);

        // LN1 -> h16
        ln64_f16<<<Mc / 8, 256, 0, stream>>>(X[br], n1w + br * 64, n1b + br * 64, h16, Mc);

        // in_proj GEMM (WMMA): [M,64] @ [64,256]; epilogue splits xc(NCHW)/silu(z)
        wmma_gemm<64, 4, 0><<<dim3(Mc / 16, 256 / 64), 32, 0, stream>>>(
            h16, inp16, nullptr, nullptr, nullptr, xc, zbuf, nullptr);

        // depthwise conv + SiLU -> both scan layouts
        conv_silu<<<(Bc * DIc * Lc) / 256, 256, 0, stream>>>(xc, convw + br * DIc * 9, x0T, x1T);

        // x_proj: dts/Bs/Cs per direction
        xdbl_kernel<<<dim3(Lc / 256, Kdir, Bc), 256, 0, stream>>>(x0T, x1T, xpw + br * Kdir * 6 * DIc, xdbl);

        // selective scan with fused direction merge (atomic add into ycomb)
        zerof<<<(Mc * DIc) / 256, 256, 0, stream>>>(ycomb, Mc * DIc);
        scan_kernel<<<Bc * Kdir, 128, 0, stream>>>(
            xdbl, x0T, x1T, dtw + br * Kdir * DIc * Rr, dtb + br * Kdir * DIc,
            alog + br * Kdir * DIc, dsv + br * Kdir * DIc, ycomb);

        // out-norm * silu(z) -> g16
        outln128_mulz<<<Mc / 8, 256, 0, stream>>>(ycomb, onw + br * DIc, onb + br * DIc, zbuf, g16, Mc);

        // out_proj GEMM (WMMA) + residual: [M,128] @ [128,64]
        wmma_gemm<128, 4, 1><<<dim3(Mc / 16, 1), 32, 0, stream>>>(
            g16, outp16, nullptr, X[br], nullptr, xres, nullptr, nullptr);

        // LN2 -> h2_16
        ln64_f16<<<Mc / 8, 256, 0, stream>>>(xres, n2w + br * 64, n2b + br * 64, h2_16, Mc);

        // fc1 GEMM (WMMA) + bias + GELU -> a16
        wmma_gemm<64, 4, 2><<<dim3(Mc / 16, 256 / 64), 32, 0, stream>>>(
            h2_16, fc1w16, fc1b + br * 256, nullptr, nullptr, nullptr, nullptr, a16);

        // fc2 GEMM (WMMA) + bias + residual; branch 1 also sums branches and
        // duplicates into both halves of the tuple output
        if (br == 0) {
            wmma_gemm<256, 4, 3><<<dim3(Mc / 16, 1), 32, 0, stream>>>(
                a16, fc2w16, fc2b + br * 64, xres, nullptr, bout, nullptr, nullptr);
        } else {
            wmma_gemm<256, 4, 4><<<dim3(Mc / 16, 1), 32, 0, stream>>>(
                a16, fc2w16, fc2b + br * 64, xres, bout, out, nullptr, nullptr);
        }
    }
}
